// MultiheadAttention_73057393705544
// MI455X (gfx1250) — matrix operand order repaired and checked by reading, not compiled, not run on hardware
//
#include <hip/hip_runtime.h>
#include <hip/hip_bf16.h>
#include <stdint.h>

#define BB 2
#define SS 2048
#define EE 1024
#define HH 16
#define DD 64

typedef __attribute__((ext_vector_type(16))) __bf16 v16bf;
typedef __attribute__((ext_vector_type(8)))  __bf16 bf16x8;
typedef __attribute__((ext_vector_type(8)))  float  v8f;

// Native conversion: gfx1250 has hardware f32->bf16 (v_cvt_pk_bf16_f32); let clang lower it.
__device__ __forceinline__ __bf16 f2bf(float f) { return (__bf16)f; }

__device__ __forceinline__ void cvt4(float4 v, __bf16* d) {
  d[0] = f2bf(v.x); d[1] = f2bf(v.y); d[2] = f2bf(v.z); d[3] = f2bf(v.w);
}

// 16 contiguous bf16 -> B-fragment (lane holds K=+0..15 of its half)
__device__ __forceinline__ v16bf ld16(const __bf16* p) {
  bf16x8 lo = *(const bf16x8*)p;
  bf16x8 hi = *(const bf16x8*)(p + 8);
  v16bf r;
#pragma unroll
  for (int i = 0; i < 8; ++i) { r[i] = lo[i]; r[i + 8] = hi[i]; }
  return r;
}

// A-fragment (bf16 source): 16 contiguous elems at p (K=+0..15)
__device__ __forceinline__ v16bf ld_afrag(const __bf16* p) {
  bf16x8 lo = *(const bf16x8*)p;
  bf16x8 hi = *(const bf16x8*)(p + 8);
  v16bf r;
#pragma unroll
  for (int i = 0; i < 8; ++i) { r[i] = lo[i]; r[i + 8] = hi[i]; }
  return r;
}

// A-fragment from fp32 source (inline convert; amortized over 4 WMMAs)
__device__ __forceinline__ v16bf ld_afrag(const float* p) {
  const float4* a = (const float4*)p;
  float4 x0 = a[0], x1 = a[1], x2 = a[2], x3 = a[3];
  __bf16 t[16];
  cvt4(x0, t); cvt4(x1, t + 4); cvt4(x2, t + 8); cvt4(x3, t + 12);
  v16bf r;
#pragma unroll
  for (int i = 0; i < 16; ++i) r[i] = t[i];
  return r;
}

#define WMMA_BF16(A, B, C) \
  __builtin_amdgcn_wmma_f32_16x16x32_bf16(false, (A), false, (B), (short)0, (C), false, false)

// One-time fp32 -> bf16 conversion (b128 in, b128 out); n must be a multiple of 8.
__global__ __launch_bounds__(256) void conv_bf16(const float* __restrict__ src,
                                                 __bf16* __restrict__ dst, int n) {
  const int i = (blockIdx.x * 256 + threadIdx.x) * 8;
  if (i >= n) return;
  const float4* s = (const float4*)(src + i);
  float4 x0 = s[0], x1 = s[1];
  __bf16 t[8];
  cvt4(x0, t); cvt4(x1, t + 4);
  bf16x8 r;
#pragma unroll
  for (int j = 0; j < 8; ++j) r[j] = t[j];
  *(bf16x8*)(dst + i) = r;
}

// C = alpha * (A @ Wb^T + bias) ; each wave computes a 16x64 tile (4 n-tiles), K = EE.
// MODE 0: bf16 out row-major [M,EE]   MODE 1: bf16 out transposed [B,H,D,S]   MODE 2: f32 out row-major
template <typename AT, int MODE>
__global__ __launch_bounds__(128) void proj_gemm(const AT* __restrict__ A,
                                                 const __bf16* __restrict__ Wb,
                                                 const float* __restrict__ bias,
                                                 __bf16* __restrict__ outb,
                                                 float* __restrict__ outf,
                                                 float alpha) {
  const int lane = threadIdx.x & 31;
  const int wid  = threadIdx.x >> 5;
  const int m0  = blockIdx.x * 16;
  const int nt0 = (blockIdx.y * 4 + wid) * 4;   // first of 4 n-tiles
  const int nn = lane & 15;
  const int hi = lane >> 4;
  const int kgA = hi * 16;
  const int kgB = hi * 16;
  const AT* arow = A + (size_t)(m0 + nn) * EE;
  const __bf16* w0 = Wb + (size_t)((nt0 + 0) * 16 + nn) * EE + kgB;
  const __bf16* w1 = Wb + (size_t)((nt0 + 1) * 16 + nn) * EE + kgB;
  const __bf16* w2 = Wb + (size_t)((nt0 + 2) * 16 + nn) * EE + kgB;
  const __bf16* w3 = Wb + (size_t)((nt0 + 3) * 16 + nn) * EE + kgB;
  v8f c0 = {}, c1 = {}, c2 = {}, c3 = {};
#pragma unroll 2
  for (int k0 = 0; k0 < EE; k0 += 32) {
    const v16bf a = ld_afrag(arow + k0 + kgA);
    c0 = WMMA_BF16(a, ld16(w0 + k0), c0);
    c1 = WMMA_BF16(a, ld16(w1 + k0), c1);
    c2 = WMMA_BF16(a, ld16(w2 + k0), c2);
    c3 = WMMA_BF16(a, ld16(w3 + k0), c3);
  }
  auto store_tile = [&](const v8f& c, int n0) {
    const float bv = bias[n0 + nn];
#pragma unroll
    for (int r = 0; r < 8; ++r) {
      const int row = m0 + r + hi * 8;
      const float val = alpha * (c[r] + bv);
      if constexpr (MODE == 0) {
        outb[(size_t)row * EE + n0 + nn] = f2bf(val);
      } else if constexpr (MODE == 1) {
        const int e = n0 + nn, h = e >> 6, d = e & 63;
        const int b_ = row >> 11, s = row & (SS - 1);
        outb[(((size_t)b_ * HH + h) * DD + d) * SS + s] = f2bf(val);
      } else {
        outf[(size_t)row * EE + n0 + nn] = val;
      }
    }
  };
  store_tile(c0, (nt0 + 0) * 16);
  store_tile(c1, (nt0 + 1) * 16);
  store_tile(c2, (nt0 + 2) * 16);
  store_tile(c3, (nt0 + 3) * 16);
}

// Flash-style causal attention: one 16-row q-tile per wave; KV tiles of 64 (16 WMMAs/tile).
// Q is pre-scaled by E^-0.5 (folded into the Q projection; exact power of two).
__global__ __launch_bounds__(128) void attn_fwd(const __bf16* __restrict__ Qb,
                                                const __bf16* __restrict__ Kb,
                                                const __bf16* __restrict__ Vt,
                                                __bf16* __restrict__ Ob) {
  __shared__ __align__(16) __bf16 pT[4][16 * 64];
  const int lane = threadIdx.x & 31;
  const int wid  = threadIdx.x >> 5;
  const int qt = blockIdx.x * 4 + wid;
  const int b  = blockIdx.y >> 4;
  const int h  = blockIdx.y & 15;
  const int q0 = qt * 16;
  const int nn = lane & 15, hi = lane >> 4;
  const int kgA = hi * 16, kgB = hi * 16;
  const size_t base = (size_t)b * SS * EE + (size_t)h * DD;

  // Q A-fragments for the two 32-wide d-steps (D=64)
  const v16bf aq0 = ld_afrag(Qb + base + (size_t)(q0 + nn) * EE + kgA);
  const v16bf aq1 = ld_afrag(Qb + base + (size_t)(q0 + nn) * EE + 32 + kgA);

  v8f o0 = {}, o1 = {}, o2 = {}, o3 = {};
  float mrow[8], lrow[8];
#pragma unroll
  for (int r = 0; r < 8; ++r) { mrow[r] = -1e30f; lrow[r] = 0.f; }

  const int ntiles = (q0 + 79) >> 6;     // causal: tiles covering columns 0..q0+15

  for (int t = 0; t < ntiles; ++t) {
    const int kt = t * 64;
    // ---- scores: 16x64 = 4 fragments, 8 WMMAs ----
    v8f s0 = {}, s1 = {}, s2 = {}, s3 = {};
    {
      const __bf16* kp0 = Kb + base + (size_t)(kt + nn) * EE;
      const __bf16* kp1 = Kb + base + (size_t)(kt + 16 + nn) * EE;
      const __bf16* kp2 = Kb + base + (size_t)(kt + 32 + nn) * EE;
      const __bf16* kp3 = Kb + base + (size_t)(kt + 48 + nn) * EE;
      s0 = WMMA_BF16(aq0, ld16(kp0 + kgB), s0);
      s1 = WMMA_BF16(aq0, ld16(kp1 + kgB), s1);
      s2 = WMMA_BF16(aq0, ld16(kp2 + kgB), s2);
      s3 = WMMA_BF16(aq0, ld16(kp3 + kgB), s3);
      s0 = WMMA_BF16(aq1, ld16(kp0 + 32 + kgB), s0);
      s1 = WMMA_BF16(aq1, ld16(kp1 + 32 + kgB), s1);
      s2 = WMMA_BF16(aq1, ld16(kp2 + 32 + kgB), s2);
      s3 = WMMA_BF16(aq1, ld16(kp3 + 32 + kgB), s3);
    }
    const bool need_mask = (kt + 63 > q0);   // wave-uniform: only diagonal tiles mask
    // ---- online softmax over 64 columns ----
#pragma unroll
    for (int r = 0; r < 8; ++r) {
      const int q = q0 + r + hi * 8;
      float x0 = s0[r], x1 = s1[r], x2 = s2[r], x3 = s3[r];
      if (need_mask) {
        x0 = (kt + nn      > q) ? -1e30f : x0;
        x1 = (kt + 16 + nn > q) ? -1e30f : x1;
        x2 = (kt + 32 + nn > q) ? -1e30f : x2;
        x3 = (kt + 48 + nn > q) ? -1e30f : x3;
      }
      float rm = fmaxf(fmaxf(x0, x1), fmaxf(x2, x3));
#pragma unroll
      for (int msk = 1; msk < 16; msk <<= 1) rm = fmaxf(rm, __shfl_xor(rm, msk, 32));
      const float mn = fmaxf(mrow[r], rm);
      const float p0 = __expf(x0 - mn);
      const float p1 = __expf(x1 - mn);
      const float p2 = __expf(x2 - mn);
      const float p3 = __expf(x3 - mn);
      float rs = (p0 + p1) + (p2 + p3);
#pragma unroll
      for (int msk = 1; msk < 16; msk <<= 1) rs += __shfl_xor(rs, msk, 32);
      const float corr = __expf(mrow[r] - mn);
      lrow[r] = lrow[r] * corr + rs;
      mrow[r] = mn;
      o0[r] *= corr; o1[r] *= corr; o2[r] *= corr; o3[r] *= corr;
      __bf16* pr = &pT[wid][(r + hi * 8) * 64 + nn];
      pr[0]  = f2bf(p0);
      pr[16] = f2bf(p1);
      pr[32] = f2bf(p2);
      pr[48] = f2bf(p3);
    }
    asm volatile("s_wait_dscnt 0" ::: "memory");
    // ---- P @ V : K-dim 64 -> 2 A-fragments, 8 WMMAs ----
    const v16bf ap0 = ld_afrag(&pT[wid][nn * 64 + kgA]);
    const v16bf ap1 = ld_afrag(&pT[wid][nn * 64 + 32 + kgA]);
    const size_t vbase = ((size_t)b * HH + h) * DD * SS + kt + kgB;
    const __bf16* vp0 = Vt + vbase + (size_t)( 0 + nn) * SS;
    const __bf16* vp1 = Vt + vbase + (size_t)(16 + nn) * SS;
    const __bf16* vp2 = Vt + vbase + (size_t)(32 + nn) * SS;
    const __bf16* vp3 = Vt + vbase + (size_t)(48 + nn) * SS;
    o0 = WMMA_BF16(ap0, ld16(vp0), o0);
    o1 = WMMA_BF16(ap0, ld16(vp1), o1);
    o2 = WMMA_BF16(ap0, ld16(vp2), o2);
    o3 = WMMA_BF16(ap0, ld16(vp3), o3);
    o0 = WMMA_BF16(ap1, ld16(vp0 + 32), o0);
    o1 = WMMA_BF16(ap1, ld16(vp1 + 32), o1);
    o2 = WMMA_BF16(ap1, ld16(vp2 + 32), o2);
    o3 = WMMA_BF16(ap1, ld16(vp3 + 32), o3);
    asm volatile("s_wait_dscnt 0" ::: "memory");
  }

#pragma unroll
  for (int r = 0; r < 8; ++r) {
    const float inv = 1.0f / lrow[r];
    __bf16* op = Ob + ((size_t)b * SS + q0 + r + hi * 8) * EE + (size_t)h * DD + nn;
    op[0]  = f2bf(o0[r] * inv);
    op[16] = f2bf(o1[r] * inv);
    op[32] = f2bf(o2[r] * inv);
    op[48] = f2bf(o3[r] * inv);
  }
}

extern "C" void kernel_launch(void* const* d_in, const int* in_sizes, int n_in,
                              void* d_out, int out_size, void* d_ws, size_t ws_size,
                              hipStream_t stream) {
  (void)in_sizes; (void)n_in; (void)out_size; (void)ws_size;
  const float* q  = (const float*)d_in[0];
  const float* k  = (const float*)d_in[1];
  const float* v  = (const float*)d_in[2];
  const float* Wq = (const float*)d_in[3];
  const float* bq = (const float*)d_in[4];
  const float* Wk = (const float*)d_in[5];
  const float* bk = (const float*)d_in[6];
  const float* Wv = (const float*)d_in[7];
  const float* bv = (const float*)d_in[8];
  const float* Wl = (const float*)d_in[9];
  const float* bl = (const float*)d_in[10];
  float* out = (float*)d_out;

  const size_t act = (size_t)BB * SS * EE;   // 4M elems
  const size_t wsz = (size_t)EE * EE;        // 1M elems
  __bf16* Qb  = (__bf16*)d_ws;               // [B*S, E] bf16 (pre-scaled by 1/32)
  __bf16* Kb  = Qb + act;                    // [B*S, E] bf16
  __bf16* Vt  = Kb + act;                    // [B, H, D, S] bf16 (transposed)
  __bf16* Ob  = Vt + act;                    // [B*S, E] bf16
  __bf16* Wqb = Ob + act;                    // [E, E] bf16 weights
  __bf16* Wkb = Wqb + wsz;
  __bf16* Wvb = Wkb + wsz;
  __bf16* Wlb = Wvb + wsz;

  const dim3 cb(256), cg((unsigned)(wsz / 8 / 256));
  hipLaunchKernelGGL(conv_bf16, cg, cb, 0, stream, Wq, Wqb, (int)wsz);
  hipLaunchKernelGGL(conv_bf16, cg, cb, 0, stream, Wk, Wkb, (int)wsz);
  hipLaunchKernelGGL(conv_bf16, cg, cb, 0, stream, Wv, Wvb, (int)wsz);
  hipLaunchKernelGGL(conv_bf16, cg, cb, 0, stream, Wl, Wlb, (int)wsz);

  const dim3 pb(128);
  const dim3 pg(BB * SS / 16, EE / 16 / 16);   // (256, 4): 4 waves x 4 n-tiles per block
  hipLaunchKernelGGL((proj_gemm<float, 0>),  pg, pb, 0, stream, q, Wqb, bq, Qb, (float*)nullptr, 0.03125f);
  hipLaunchKernelGGL((proj_gemm<float, 0>),  pg, pb, 0, stream, k, Wkb, bk, Kb, (float*)nullptr, 1.0f);
  hipLaunchKernelGGL((proj_gemm<float, 1>),  pg, pb, 0, stream, v, Wvb, bv, Vt, (float*)nullptr, 1.0f);
  hipLaunchKernelGGL(attn_fwd, dim3(SS / 64, BB * HH), pb, 0, stream, Qb, Kb, Vt, Ob);
  hipLaunchKernelGGL((proj_gemm<__bf16, 2>), pg, pb, 0, stream, Ob, Wlb, bl, (__bf16*)nullptr, out, 1.0f);
}
